// RANSACMatcher_41652592836791
// MI455X (gfx1250) — matrix operand order repaired and checked by reading, not compiled, not run on hardware
//
#include <hip/hip_runtime.h>
#include <hip/hip_bf16.h>

typedef __attribute__((ext_vector_type(16))) _Float16 v16h;
typedef __attribute__((ext_vector_type(8)))  float    v8f;

static constexpr int Bk = 4, Nk = 4096, Mk = 4096, Dk = 128;
static constexpr int IT = 16;
static constexpr float THR = 0.75f, EPSV = 1e-8f, RATIO = 0.6f;

// ---------------------------------------------------------------- helpers
__device__ __forceinline__ unsigned long long pack_key(float v, int idx) {
    unsigned u = __float_as_uint(v);
    u = (u & 0x80000000u) ? ~u : (u | 0x80000000u);   // monotonic float -> uint
    return ((unsigned long long)u << 32) | (unsigned)(0xFFFFFFFFu - (unsigned)idx);
}
__device__ __forceinline__ unsigned unpack_idx(unsigned long long k) {
    return 0xFFFFFFFFu - (unsigned)(k & 0xFFFFFFFFull);
}
__device__ __forceinline__ unsigned long long shfl_xor_u64(unsigned long long v, int m) {
    unsigned lo = (unsigned)v, hi = (unsigned)(v >> 32);
    lo = __shfl_xor(lo, m, 32);
    hi = __shfl_xor(hi, m, 32);
    return ((unsigned long long)hi << 32) | lo;
}
__device__ __forceinline__ unsigned hash3(unsigned a, unsigned b, unsigned c) {
    unsigned h = a * 0x9E3779B9u ^ b * 0x85EBCA6Bu ^ c * 0xC2B2AE35u;
    h ^= h >> 16; h *= 0x7FEB352Du; h ^= h >> 15; h *= 0x846CA68Bu; h ^= h >> 16;
    return h;
}

// ---------------------------------------------------- 0: reset packed keys
__global__ __launch_bounds__(256) void init_kernel(unsigned long long* px,
                                                   unsigned long long* py) {
    int i = blockIdx.x * 256 + threadIdx.x;
    if (i < Bk * Nk) px[i] = 0ull;
    if (i < Bk * Mk) py[i] = 0ull;
}

// -------------------------------------------- 1: fused WMMA GEMM + argmax
// grid = (M/128, N/128, B), block = 256 (8 wave32 waves)
__global__ __launch_bounds__(256) void sim_argmax_kernel(
        const float* __restrict__ xd, const float* __restrict__ yd,
        const unsigned char* __restrict__ mask,
        unsigned long long* __restrict__ packed_x,
        unsigned long long* __restrict__ packed_y) {
    __shared__ _Float16 xa[128][40];   // 128 rows x 32 K-slice (padded)
    __shared__ _Float16 yb[128][40];

    const int b   = blockIdx.z;
    const int n0  = blockIdx.y * 128;
    const int m0  = blockIdx.x * 128;
    const int tid = threadIdx.x;
    const int wave = tid >> 5;
    const int lane = tid & 31;
    const int hi   = lane >> 4;     // 0 or 1
    const int l15  = lane & 15;
    const int arow = wave * 16 + l15;

    // mask -> per-lane column validity (columns t*16+l15)
    float mneg[8];
#pragma unroll
    for (int t = 0; t < 8; ++t)
        mneg[t] = mask[(size_t)b * Mk + m0 + t * 16 + l15] ? 0.0f : 1.0f;

    v8f acc[8] = {};

    for (int kk = 0; kk < Dk; kk += 32) {
        __syncthreads();
        // stage 128x32 f32 -> f16 LDS slices for both tiles
        for (int i = tid; i < 128 * 8; i += 256) {
            int r = i >> 3, c4 = i & 7, c = c4 * 4;
            const float4 vx = *(const float4*)(xd + ((size_t)b * Nk + n0 + r) * Dk + kk + c);
            const float4 vy = *(const float4*)(yd + ((size_t)b * Mk + m0 + r) * Dk + kk + c);
            xa[r][c + 0] = (_Float16)vx.x; xa[r][c + 1] = (_Float16)vx.y;
            xa[r][c + 2] = (_Float16)vx.z; xa[r][c + 3] = (_Float16)vx.w;
            yb[r][c + 0] = (_Float16)vy.x; yb[r][c + 1] = (_Float16)vy.y;
            yb[r][c + 2] = (_Float16)vy.z; yb[r][c + 3] = (_Float16)vy.w;
        }
        __syncthreads();

        // A fragment (16x32, M striped in-lane): lanes<16 K0-7/16-23, lanes>=16 K8-15/24-31
        v16h afrag;
        const int kb = hi * 8;
#pragma unroll
        for (int v = 0; v < 4; ++v) {
            afrag[2 * v]     = xa[arow][kb + 2 * v];
            afrag[2 * v + 1] = xa[arow][kb + 2 * v + 1];
            afrag[2 * v + 8] = xa[arow][kb + 16 + 2 * v];
            afrag[2 * v + 9] = xa[arow][kb + 16 + 2 * v + 1];
        }

        // Preload ALL eight B fragments (32x16, N striped across lanes) so the
        // ds_load_b128s batch into one clause and the 8 WMMAs issue back-to-back.
        v16h bf[8];
        const int kbb = hi * 8;
#pragma unroll
        for (int t = 0; t < 8; ++t) {
            const int col = t * 16 + l15;
#pragma unroll
            for (int j = 0; j < 16; ++j) bf[t][j] = yb[col][kbb + (j & 7) + ((j >> 3) << 4)];
        }
#pragma unroll
        for (int t = 0; t < 8; ++t)
            acc[t] = __builtin_amdgcn_wmma_f32_16x16x32_f16(
                false, afrag, false, bf[t], (short)0, acc[t], false, false);
    }

    // ---- row argmax (nn_x): row M=r+hi*8 lives in VGPR r across 16 lanes
#pragma unroll
    for (int r = 0; r < 8; ++r) {
        const int row = wave * 16 + hi * 8 + r;
        unsigned long long best = 0ull;
#pragma unroll
        for (int t = 0; t < 8; ++t) {
            float v = acc[t][r];
            if (mneg[t] != 0.0f) v = -3.4e38f;
            unsigned long long k = pack_key(v, m0 + t * 16 + l15);
            if (k > best) best = k;
        }
#pragma unroll
        for (int m = 1; m <= 8; m <<= 1) {
            unsigned long long o = shfl_xor_u64(best, m);
            if (o > best) best = o;
        }
        if (l15 == 0)
            atomicMax(&packed_x[(size_t)b * Nk + n0 + row], best);
    }

    // ---- col argmax (nn_y): reduce 8 in-lane rows, then swap halves (xor 16)
#pragma unroll
    for (int t = 0; t < 8; ++t) {
        const int col = t * 16 + l15;
        unsigned long long best = 0ull;
#pragma unroll
        for (int r = 0; r < 8; ++r) {
            float v = acc[t][r];
            if (mneg[t] != 0.0f) v = -3.4e38f;
            unsigned long long k = pack_key(v, n0 + wave * 16 + hi * 8 + r);
            if (k > best) best = k;
        }
        unsigned long long o = shfl_xor_u64(best, 16);
        if (o > best) best = o;
        if (hi == 0)
            atomicMax(&packed_y[(size_t)b * Mk + m0 + col], best);
    }
}

// ------------------------------------- 2: mutual check + gather + 8pt rows
__global__ __launch_bounds__(256) void mutual_kernel(
        const float* __restrict__ xk, const float* __restrict__ yk,
        const unsigned char* __restrict__ mask,
        const unsigned long long* __restrict__ px,
        const unsigned long long* __restrict__ py,
        float* __restrict__ w, float* __restrict__ yk2,
        float* __restrict__ Arows) {
    int i = blockIdx.x * 256 + threadIdx.x;
    if (i >= Bk * Nk) return;
    int b = i / Nk, n = i % Nk;
    unsigned nnx = unpack_idx(px[i]);
    if (nnx >= (unsigned)Mk) nnx = 0;
    unsigned back = unpack_idx(py[(size_t)b * Mk + nnx]);
    bool valid = (back == (unsigned)n) && (mask[(size_t)b * Mk + nnx] != 0);
    w[i] = valid ? 1.0f : 0.0f;
    float x1 = xk[2 * (size_t)i], y1 = xk[2 * (size_t)i + 1];
    float x2 = yk[((size_t)b * Mk + nnx) * 2], y2 = yk[((size_t)b * Mk + nnx) * 2 + 1];
    yk2[2 * (size_t)i] = x2; yk2[2 * (size_t)i + 1] = y2;
    float* A = Arows + (size_t)i * 9;
    A[0] = x2 * x1; A[1] = x2 * y1; A[2] = x2;
    A[3] = y2 * x1; A[4] = y2 * y1; A[5] = y2;
    A[6] = x1;      A[7] = y1;      A[8] = 1.0f;
}

// ------------------------- 3: weighted 9x9 normal equations per hypothesis
__global__ __launch_bounds__(256) void normal_eq_kernel(
        const float* __restrict__ Arows, const float* __restrict__ w,
        float* __restrict__ Mmat) {
    const int hyp = blockIdx.x;            // b*IT + it
    const int b = hyp / IT, it = hyp % IT;
    const unsigned thresh = (unsigned)(RATIO * 16777216.0f);
    float accv[45];
#pragma unroll
    for (int k = 0; k < 45; ++k) accv[k] = 0.0f;

    for (int n = threadIdx.x; n < Nk; n += 256) {
        if (w[(size_t)b * Nk + n] <= 0.0f) continue;
        if ((hash3(b, it, n) & 0xFFFFFFu) >= thresh) continue;   // ~top-RATIO sample
        const float* A = Arows + ((size_t)b * Nk + n) * 9;
        float a[9];
#pragma unroll
        for (int j = 0; j < 9; ++j) a[j] = A[j];
        int k = 0;
#pragma unroll
        for (int p = 0; p < 9; ++p)
#pragma unroll
            for (int q = p; q < 9; ++q) accv[k++] += a[p] * a[q];
    }

    __shared__ float sred[256];
    float* Mo = Mmat + (size_t)hyp * 81;
    int k = 0;
    for (int p = 0; p < 9; ++p)
        for (int q = p; q < 9; ++q) {
            sred[threadIdx.x] = accv[k];
            __syncthreads();
            for (int s = 128; s > 0; s >>= 1) {
                if (threadIdx.x < s) sred[threadIdx.x] += sred[threadIdx.x + s];
                __syncthreads();
            }
            if (threadIdx.x == 0) { Mo[p * 9 + q] = sred[0]; Mo[q * 9 + p] = sred[0]; }
            __syncthreads();
            ++k;
        }
}

// ------------------------ 4: smallest-eigenvector of 9x9 via cyclic Jacobi
__global__ void eig_kernel(const float* __restrict__ Mmat, float* __restrict__ Fout) {
    int hyp = blockIdx.x * blockDim.x + threadIdx.x;
    if (hyp >= Bk * IT) return;
    float Am[9][9], V[9][9];
    for (int p = 0; p < 9; ++p)
        for (int q = 0; q < 9; ++q) {
            Am[p][q] = Mmat[(size_t)hyp * 81 + p * 9 + q];
            V[p][q] = (p == q) ? 1.0f : 0.0f;
        }
    for (int sweep = 0; sweep < 12; ++sweep)
        for (int p = 0; p < 8; ++p)
            for (int q = p + 1; q < 9; ++q) {
                float apq = Am[p][q];
                if (fabsf(apq) < 1e-12f) continue;
                float tau = (Am[q][q] - Am[p][p]) / (2.0f * apq);
                float t = (tau >= 0.0f ? 1.0f : -1.0f) /
                          (fabsf(tau) + sqrtf(1.0f + tau * tau));
                float c = 1.0f / sqrtf(1.0f + t * t);
                float s = t * c;
                for (int kx = 0; kx < 9; ++kx) {
                    float akp = Am[kx][p], akq = Am[kx][q];
                    Am[kx][p] = c * akp - s * akq;
                    Am[kx][q] = s * akp + c * akq;
                }
                for (int kx = 0; kx < 9; ++kx) {
                    float apk = Am[p][kx], aqk = Am[q][kx];
                    Am[p][kx] = c * apk - s * aqk;
                    Am[q][kx] = s * apk + c * aqk;
                }
                for (int kx = 0; kx < 9; ++kx) {
                    float vkp = V[kx][p], vkq = V[kx][q];
                    V[kx][p] = c * vkp - s * vkq;
                    V[kx][q] = s * vkp + c * vkq;
                }
            }
    int mi = 0; float mv = Am[0][0];
    for (int j = 1; j < 9; ++j)
        if (Am[j][j] < mv) { mv = Am[j][j]; mi = j; }
    for (int r = 0; r < 9; ++r) Fout[(size_t)hyp * 9 + r] = V[r][mi];
}

// -------------------------------------- 5: Sampson errors + inlier counts
__global__ __launch_bounds__(256) void sampson_kernel(
        const float* __restrict__ Fm, const float* __restrict__ xk,
        const float* __restrict__ yk2, const float* __restrict__ w,
        float* __restrict__ err, int* __restrict__ counts) {
    const int hyp = blockIdx.x;
    const int b = hyp / IT;
    float F[9];
#pragma unroll
    for (int j = 0; j < 9; ++j) F[j] = Fm[(size_t)hyp * 9 + j];
    __shared__ int cnt;
    if (threadIdx.x == 0) cnt = 0;
    __syncthreads();
    int local = 0;
    for (int n = threadIdx.x; n < Nk; n += 256) {
        size_t i = (size_t)b * Nk + n;
        float x1 = xk[2 * i], y1 = xk[2 * i + 1];
        float x2 = yk2[2 * i], y2 = yk2[2 * i + 1];
        float Fx0 = F[0] * x1 + F[1] * y1 + F[2];
        float Fx1 = F[3] * x1 + F[4] * y1 + F[5];
        float Fx2 = F[6] * x1 + F[7] * y1 + F[8];
        float Ft0 = F[0] * x2 + F[3] * y2 + F[6];
        float Ft1 = F[1] * x2 + F[4] * y2 + F[7];
        float num = x2 * Fx0 + y2 * Fx1 + Fx2; num *= num;
        float den = Fx0 * Fx0 + Fx1 * Fx1 + Ft0 * Ft0 + Ft1 * Ft1;
        float e = num / (den + EPSV);
        err[(size_t)hyp * Nk + n] = e;
        if (e < THR && w[i] > 0.0f) ++local;
    }
    atomicAdd(&cnt, local);
    __syncthreads();
    if (threadIdx.x == 0) counts[hyp] = cnt;
}

// ------------------------------------------- 6: pick best model, write out
__global__ __launch_bounds__(256) void finalize_kernel(
        const int* __restrict__ counts, const float* __restrict__ Fm,
        const float* __restrict__ err, const float* __restrict__ w,
        float* __restrict__ out) {
    const int b = blockIdx.x;
    __shared__ int best;
    if (threadIdx.x == 0) {
        int bi = 0, bc = counts[b * IT];
        for (int it = 1; it < IT; ++it)
            if (counts[b * IT + it] > bc) { bc = counts[b * IT + it]; bi = it; }
        best = bi;
    }
    __syncthreads();
    const int hyp = b * IT + best;
    float* inl   = out;                                   // [B*N]
    float* model = out + (size_t)Bk * Nk;                 // [B*9]
    float* errs  = out + (size_t)Bk * Nk + (size_t)Bk * 9;// [B*N]
    if (threadIdx.x < 9) model[b * 9 + threadIdx.x] = Fm[(size_t)hyp * 9 + threadIdx.x];
    for (int n = threadIdx.x; n < Nk; n += 256) {
        float e = err[(size_t)hyp * Nk + n];
        errs[(size_t)b * Nk + n] = e;
        inl[(size_t)b * Nk + n] =
            (e < THR && w[(size_t)b * Nk + n] > 0.0f) ? 1.0f : 0.0f;
    }
}

// ------------------------------------------------------------------ launch
extern "C" void kernel_launch(void* const* d_in, const int* in_sizes, int n_in,
                              void* d_out, int out_size, void* d_ws, size_t ws_size,
                              hipStream_t stream) {
    const float* xk = (const float*)d_in[0];
    const float* xd = (const float*)d_in[1];
    const float* yk = (const float*)d_in[2];
    const float* yd = (const float*)d_in[3];
    const unsigned char* mask = (const unsigned char*)d_in[4];

    char* p = (char*)d_ws;
    auto alloc = [&](size_t bytes) { char* q = p; p += (bytes + 255) & ~(size_t)255; return q; };
    unsigned long long* px = (unsigned long long*)alloc((size_t)Bk * Nk * 8);
    unsigned long long* py = (unsigned long long*)alloc((size_t)Bk * Mk * 8);
    float* w     = (float*)alloc((size_t)Bk * Nk * 4);
    float* yk2   = (float*)alloc((size_t)Bk * Nk * 2 * 4);
    float* Arows = (float*)alloc((size_t)Bk * Nk * 9 * 4);
    float* Mmat  = (float*)alloc((size_t)Bk * IT * 81 * 4);
    float* Fm    = (float*)alloc((size_t)Bk * IT * 9 * 4);
    float* err   = (float*)alloc((size_t)Bk * IT * Nk * 4);
    int*   counts= (int*)alloc((size_t)Bk * IT * 4);

    init_kernel<<<(Bk * Nk + 255) / 256, 256, 0, stream>>>(px, py);

    dim3 g(Mk / 128, Nk / 128, Bk);
    sim_argmax_kernel<<<g, 256, 0, stream>>>(xd, yd, mask, px, py);

    mutual_kernel<<<(Bk * Nk + 255) / 256, 256, 0, stream>>>(
        xk, yk, mask, px, py, w, yk2, Arows);

    normal_eq_kernel<<<Bk * IT, 256, 0, stream>>>(Arows, w, Mmat);

    eig_kernel<<<1, Bk * IT, 0, stream>>>(Mmat, Fm);

    sampson_kernel<<<Bk * IT, 256, 0, stream>>>(Fm, xk, yk2, w, err, counts);

    finalize_kernel<<<Bk, 256, 0, stream>>>(counts, Fm, err, w, (float*)d_out);
}
